// MultiHeadAttention_63178968924635
// MI455X (gfx1250) — compile-verified, matrix operand order checked by reading, not run on hardware
//
#include <hip/hip_runtime.h>

#define HEADS   8
#define DM      512
#define DK      64
#define BATCH   2
#define SEQ     2048
#define PROW    2052   // padded LDS row stride (floats): stagger banks across rows

// ---- plain vector types (trivially constructible, safe in unions) ----
typedef __attribute__((ext_vector_type(16))) __bf16      v16bf;
typedef __attribute__((ext_vector_type(8)))  float       v8f;
typedef __attribute__((ext_vector_type(4)))  unsigned int u32x4;
typedef __attribute__((ext_vector_type(4)))  float       f32x4;

union Frag {
    v16bf          v;      // 16 bf16 = 8 VGPRs
    u32x4          u[2];   // two 16-byte chunks
    unsigned short e[16];
};

__device__ __forceinline__ unsigned short f32_to_bf16(float f) {
    unsigned int u = __float_as_uint(f);
    unsigned int r = u + 0x7FFFu + ((u >> 16) & 1u);   // round-to-nearest-even
    return (unsigned short)(r >> 16);
}

__device__ __forceinline__ u32x4 ld_b128(const unsigned short* p) {
    return *reinterpret_cast<const u32x4*>(p);
}

__device__ __forceinline__ v8f wmma_bf16(Frag a, Frag b, v8f c) {
    return __builtin_amdgcn_wmma_f32_16x16x32_bf16(false, a.v, false, b.v,
                                                   (short)0, c, false, false);
}

__device__ __forceinline__ void cvt_frag_f32(Frag& a, const float* pa) {
    f32x4 f0 = *reinterpret_cast<const f32x4*>(pa);
    f32x4 f1 = *reinterpret_cast<const f32x4*>(pa + 4);
    f32x4 f2 = *reinterpret_cast<const f32x4*>(pa + 16);
    f32x4 f3 = *reinterpret_cast<const f32x4*>(pa + 20);
    a.e[0]=f32_to_bf16(f0.x);  a.e[1]=f32_to_bf16(f0.y);
    a.e[2]=f32_to_bf16(f0.z);  a.e[3]=f32_to_bf16(f0.w);
    a.e[4]=f32_to_bf16(f1.x);  a.e[5]=f32_to_bf16(f1.y);
    a.e[6]=f32_to_bf16(f1.z);  a.e[7]=f32_to_bf16(f1.w);
    a.e[8]=f32_to_bf16(f2.x);  a.e[9]=f32_to_bf16(f2.y);
    a.e[10]=f32_to_bf16(f2.z); a.e[11]=f32_to_bf16(f2.w);
    a.e[12]=f32_to_bf16(f3.x); a.e[13]=f32_to_bf16(f3.y);
    a.e[14]=f32_to_bf16(f3.z); a.e[15]=f32_to_bf16(f3.w);
}

// ---------------- prep kernels ----------------
__global__ void k_cvt_bf16(const float* __restrict__ src,
                           unsigned short* __restrict__ dst, int n) {
    int i = blockIdx.x * blockDim.x + threadIdx.x;
    if (i < n) dst[i] = f32_to_bf16(src[i]);
}

// W [K=512, N=512] f32 row-major  ->  WT [N=512, K=512] bf16 row-major
__global__ void k_transpose_bf16(const float* __restrict__ W,
                                 unsigned short* __restrict__ WT) {
    int n = blockIdx.x * 16 + threadIdx.x;
    int k = blockIdx.y * 16 + threadIdx.y;
    WT[n * DM + k] = f32_to_bf16(W[k * DM + n]);
}

// ---------------- GEMM: QKV projection (NB=4 register blocking) ----------------
// X [4096,512] bf16; WT [512,512] bf16 (WT[n][k]); bias f32[512]
// mode 0: Y[bh][s][dk] bf16 (Q,K layout)   mode 1: Y[bh][dk][s] bf16 (V transposed)
__global__ void k_gemm_proj(const unsigned short* __restrict__ X,
                            const unsigned short* __restrict__ WT,
                            const float* __restrict__ bias,
                            unsigned short* __restrict__ Y, int mode) {
    const int lane = threadIdx.x;
    const int n0 = blockIdx.x * 64;
    const int m0 = blockIdx.y * 16;
    const int koff = (lane & 16) ? 8 : 0;
    const int rowA = m0 + (lane & 15);
    const int rb   = n0 + (lane & 15);

    v8f acc[4] = {{}, {}, {}, {}};
    for (int k0 = 0; k0 < DM; k0 += 32) {
        Frag a;
        a.u[0] = ld_b128(X + rowA * DM + k0 + koff);
        a.u[1] = ld_b128(X + rowA * DM + k0 + koff + 16);
#pragma unroll
        for (int j = 0; j < 4; ++j) {
            Frag bfr;
            bfr.u[0] = ld_b128(WT + (rb + j * 16) * DM + k0 + koff);
            bfr.u[1] = ld_b128(WT + (rb + j * 16) * DM + k0 + koff + 16);
            acc[j] = wmma_bf16(a, bfr, acc[j]);
        }
    }
#pragma unroll
    for (int j = 0; j < 4; ++j) {
        const int nCol = n0 + j * 16 + (lane & 15);
        const int h  = nCol >> 6;
        const int dk = nCol & 63;
        const float bvv = bias[nCol];
#pragma unroll
        for (int r = 0; r < 8; ++r) {
            int m = m0 + r + ((lane >> 4) << 3);
            int bidx = m >> 11;
            int s = m & (SEQ - 1);
            unsigned short val = f32_to_bf16(acc[j][r] + bvv);
            if (mode == 0)
                Y[(((size_t)(bidx * HEADS + h) * SEQ) + s) * DK + dk] = val;
            else
                Y[(((size_t)(bidx * HEADS + h) * DK) + dk) * SEQ + s] = val;
        }
    }
}

// ---------------- fused attention: scores + mask + softmax + ctx ----------------
// One 256-thread (8-wave) block per (q-tile, bh). Row-block of 16x2048 scores in LDS.
__global__ void k_attn_fused(const unsigned short* __restrict__ Qh,   // [bh][s][dk]
                             const unsigned short* __restrict__ Kh,   // [bh][s][dk]
                             const unsigned short* __restrict__ VhT,  // [bh][dk][s]
                             const int* __restrict__ mask,            // [B][S][S]
                             float* __restrict__ attn,                // [bh][S][S]
                             unsigned short* __restrict__ ctxbf) {    // [B][S][512]
    extern __shared__ float smem[];
    float* probs  = smem;               // 16 x PROW
    float* ctxacc = smem + 16 * PROW;   // 16 x DK

    const int t    = threadIdx.x;
    const int wave = t >> 5;
    const int lane = t & 31;
    const int m0 = blockIdx.x * 16;
    const int bh = blockIdx.y;
    const int b  = bh >> 3;
    const int h  = bh & 7;

    const unsigned short* Q = Qh  + (size_t)bh * SEQ * DK;
    const unsigned short* K = Kh  + (size_t)bh * SEQ * DK;
    const unsigned short* V = VhT + (size_t)bh * DK * SEQ;

    for (int i = t; i < 16 * DK; i += 256) ctxacc[i] = 0.0f;

    const int koff = (lane & 16) ? 8 : 0;
    const int rowA = m0 + (lane & 15);

    // ---- Phase 1: scores for columns [wave*256, wave*256+256), Q frags loaded once ----
    Frag qa0, qa1;
    qa0.u[0] = ld_b128(Q + rowA * DK + 0  + koff);
    qa0.u[1] = ld_b128(Q + rowA * DK + 0  + koff + 16);
    qa1.u[0] = ld_b128(Q + rowA * DK + 32 + koff);
    qa1.u[1] = ld_b128(Q + rowA * DK + 32 + koff + 16);

    for (int tile = 0; tile < 16; ++tile) {
        const int n0 = wave * 256 + tile * 16;
        const int rowB = n0 + (lane & 15);
        Frag b0, b1;
        b0.u[0] = ld_b128(K + rowB * DK + 0  + koff);
        b0.u[1] = ld_b128(K + rowB * DK + 0  + koff + 16);
        b1.u[0] = ld_b128(K + rowB * DK + 32 + koff);
        b1.u[1] = ld_b128(K + rowB * DK + 32 + koff + 16);
        v8f acc = {};
        acc = wmma_bf16(qa0, b0, acc);
        acc = wmma_bf16(qa1, b1, acc);
        const int kcol = n0 + (lane & 15);
#pragma unroll
        for (int r = 0; r < 8; ++r) {
            int row = r + ((lane >> 4) << 3);
            int q = m0 + row;
            float s = acc[r] * 0.125f;                      // 1/sqrt(64)
            if (mask[((size_t)b * SEQ + q) * SEQ + kcol] == 0) s = -10000.0f;
            probs[row * PROW + kcol] = s;
        }
    }
    __syncthreads();

    // ---- Phase 2: softmax; wave w owns rows 2w, 2w+1; write normalized attn ----
    for (int rr = 0; rr < 2; ++rr) {
        const int row = wave * 2 + rr;
        float m = -3.4e38f;
        for (int i = 0; i < 64; ++i)
            m = fmaxf(m, probs[row * PROW + i * 32 + lane]);
#pragma unroll
        for (int off = 16; off > 0; off >>= 1)
            m = fmaxf(m, __shfl_xor(m, off, 32));
        float sum = 0.0f;
        for (int i = 0; i < 64; ++i) {
            float e = __expf(probs[row * PROW + i * 32 + lane] - m);
            probs[row * PROW + i * 32 + lane] = e;
            sum += e;
        }
#pragma unroll
        for (int off = 16; off > 0; off >>= 1)
            sum += __shfl_xor(sum, off, 32);
        const float inv = 1.0f / sum;
        float* gout = attn + ((size_t)bh * SEQ + (m0 + row)) * SEQ;
        for (int i = 0; i < 64; ++i) {
            float p = probs[row * PROW + i * 32 + lane] * inv;
            probs[row * PROW + i * 32 + lane] = p;
            gout[i * 32 + lane] = p;                        // coalesced 128B stores
        }
    }
    __syncthreads();

    // ---- Phase 3: ctx partials; wave w covers K range [wave*256, +256) ----
    v8f cacc[4] = {{}, {}, {}, {}};
    for (int kk = 0; kk < 8; ++kk) {
        const int kbase = wave * 256 + kk * 32;
        Frag a;
        cvt_frag_f32(a, &probs[(lane & 15) * PROW + kbase + koff]);
#pragma unroll
        for (int j = 0; j < 4; ++j) {
            const int rowB = j * 16 + (lane & 15);
            Frag bfr;
            bfr.u[0] = ld_b128(V + rowB * SEQ + kbase + koff);
            bfr.u[1] = ld_b128(V + rowB * SEQ + kbase + koff + 16);
            cacc[j] = wmma_bf16(a, bfr, cacc[j]);
        }
    }
#pragma unroll
    for (int j = 0; j < 4; ++j) {
#pragma unroll
        for (int r = 0; r < 8; ++r) {
            int row = r + ((lane >> 4) << 3);
            atomicAdd(&ctxacc[row * DK + j * 16 + (lane & 15)], cacc[j][r]); // ds_add_f32
        }
    }
    __syncthreads();

    // ---- Phase 4: emit ctx row-block as bf16 in concat layout ----
    for (int i = t; i < 16 * DK; i += 256) {
        int row = i >> 6;
        int d   = i & 63;
        ctxbf[((size_t)(b * SEQ + m0 + row)) * DM + h * DK + d] = f32_to_bf16(ctxacc[i]);
    }
}

// ---------------- out = ctx @ Wo + bo (NB=4) ----------------
__global__ void k_gemm_out(const unsigned short* __restrict__ X,   // ctx bf16 [4096,512]
                           const unsigned short* __restrict__ WT,  // WoT bf16
                           const float* __restrict__ bias,
                           float* __restrict__ out) {
    const int lane = threadIdx.x;
    const int n0 = blockIdx.x * 64;
    const int m0 = blockIdx.y * 16;
    const int koff = (lane & 16) ? 8 : 0;
    const int rowA = m0 + (lane & 15);
    const int rb   = n0 + (lane & 15);

    v8f acc[4] = {{}, {}, {}, {}};
    for (int k0 = 0; k0 < DM; k0 += 32) {
        Frag a;
        a.u[0] = ld_b128(X + rowA * DM + k0 + koff);
        a.u[1] = ld_b128(X + rowA * DM + k0 + koff + 16);
#pragma unroll
        for (int j = 0; j < 4; ++j) {
            Frag bfr;
            bfr.u[0] = ld_b128(WT + (rb + j * 16) * DM + k0 + koff);
            bfr.u[1] = ld_b128(WT + (rb + j * 16) * DM + k0 + koff + 16);
            acc[j] = wmma_bf16(a, bfr, acc[j]);
        }
    }
#pragma unroll
    for (int j = 0; j < 4; ++j) {
        const int nCol = n0 + j * 16 + (lane & 15);
        const float bvv = bias[nCol];
#pragma unroll
        for (int r = 0; r < 8; ++r) {
            int m = m0 + r + ((lane >> 4) << 3);
            out[(size_t)m * DM + nCol] = acc[j][r] + bvv;
        }
    }
}

// ---------------- host launcher ----------------
extern "C" void kernel_launch(void* const* d_in, const int* in_sizes, int n_in,
                              void* d_out, int out_size, void* d_ws, size_t ws_size,
                              hipStream_t stream) {
    (void)in_sizes; (void)n_in; (void)out_size; (void)ws_size;
    const float* q   = (const float*)d_in[0];
    const float* k   = (const float*)d_in[1];
    const float* v   = (const float*)d_in[2];
    const int*   msk = (const int*)  d_in[3];
    const float* Wq  = (const float*)d_in[4];
    const float* bq  = (const float*)d_in[5];
    const float* Wk  = (const float*)d_in[6];
    const float* bk  = (const float*)d_in[7];
    const float* Wv  = (const float*)d_in[8];
    const float* bv  = (const float*)d_in[9];
    const float* Wo  = (const float*)d_in[10];
    const float* bo  = (const float*)d_in[11];

    float* out  = (float*)d_out;
    float* attn = out + (size_t)BATCH * SEQ * DM;   // [B,H,S,S]

    char* ws = (char*)d_ws;
    size_t off = 0;
    auto carve = [&](size_t bytes) -> void* {
        void* p = ws + off;
        off += (bytes + 255) & ~(size_t)255;
        return p;
    };
    const size_t XBF  = (size_t)BATCH * SEQ * DM * 2;         // 4 MB
    const size_t WTSZ = (size_t)DM * DM * 2;                  // 0.5 MB
    const size_t HSZ  = (size_t)BATCH * HEADS * SEQ * DK * 2; // 4 MB
    unsigned short* qbf = (unsigned short*)carve(XBF);
    unsigned short* kbf = (unsigned short*)carve(XBF);
    unsigned short* vbf = (unsigned short*)carve(XBF);
    unsigned short* WqT = (unsigned short*)carve(WTSZ);
    unsigned short* WkT = (unsigned short*)carve(WTSZ);
    unsigned short* WvT = (unsigned short*)carve(WTSZ);
    unsigned short* WoT = (unsigned short*)carve(WTSZ);
    unsigned short* Qh  = (unsigned short*)carve(HSZ);
    unsigned short* Kh  = (unsigned short*)carve(HSZ);
    unsigned short* VhT = (unsigned short*)carve(HSZ);
    unsigned short* ctxbf = (unsigned short*)carve(XBF);

    const int nElems = BATCH * SEQ * DM;
    dim3 blk32(32, 1, 1);

    // 1) bf16 conversion of activations
    k_cvt_bf16<<<(nElems + 255) / 256, 256, 0, stream>>>(q, qbf, nElems);
    k_cvt_bf16<<<(nElems + 255) / 256, 256, 0, stream>>>(k, kbf, nElems);
    k_cvt_bf16<<<(nElems + 255) / 256, 256, 0, stream>>>(v, vbf, nElems);

    // 2) transpose + convert weights
    dim3 tgrid(DM / 16, DM / 16), tblk(16, 16);
    k_transpose_bf16<<<tgrid, tblk, 0, stream>>>(Wq, WqT);
    k_transpose_bf16<<<tgrid, tblk, 0, stream>>>(Wk, WkT);
    k_transpose_bf16<<<tgrid, tblk, 0, stream>>>(Wv, WvT);
    k_transpose_bf16<<<tgrid, tblk, 0, stream>>>(Wo, WoT);

    // 3) Q/K/V projections (head-split outputs), NB=4 blocking
    dim3 pgrid(DM / 64, (BATCH * SEQ) / 16, 1);
    k_gemm_proj<<<pgrid, blk32, 0, stream>>>(qbf, WqT, bq, Qh, 0);
    k_gemm_proj<<<pgrid, blk32, 0, stream>>>(kbf, WkT, bk, Kh, 0);
    k_gemm_proj<<<pgrid, blk32, 0, stream>>>(vbf, WvT, bv, VhT, 1);

    // 4) fused scores + mask + softmax + ctx (132 KB dynamic LDS per workgroup)
    dim3 fgrid(SEQ / 16, BATCH * HEADS);
    size_t smemBytes = (size_t)(16 * PROW + 16 * DK) * sizeof(float);
    k_attn_fused<<<fgrid, 256, smemBytes, stream>>>(Qh, Kh, VhT, msk, attn, ctxbf);

    // 5) output projection, NB=4 blocking
    dim3 ogrid(DM / 64, (BATCH * SEQ) / 16, 1);
    k_gemm_out<<<ogrid, blk32, 0, stream>>>(ctxbf, WoT, bo, out);
}
